// CosineSimilarity_39256001085989
// MI455X (gfx1250) — matrix operand order repaired and checked by reading, not compiled, not run on hardware
//
#include <hip/hip_runtime.h>
#include <hip/hip_bf16.h>
#include <math.h>

typedef __attribute__((ext_vector_type(16))) __bf16 v16bf;
typedef __attribute__((ext_vector_type(8)))  __bf16 v8bf;
typedef __attribute__((ext_vector_type(8)))  float  v8f;
typedef __attribute__((ext_vector_type(4)))  float  f32x4;

#define TILE_K 32
#define BM     128
#define BN     128
#define LDSW   40   // bf16 per LDS row: 40*2 = 80 B, multiple of 16 -> b128-aligned, staggers banks

// ---------------------------------------------------------------------------
// Row L2 norms: one wave32 per row.
// ---------------------------------------------------------------------------
__global__ void rownorm_kernel(const float* __restrict__ x, float* __restrict__ out,
                               int rows, int d) {
    int gtid = blockIdx.x * blockDim.x + threadIdx.x;
    int row  = gtid >> 5;
    int lane = threadIdx.x & 31;
    if (row >= rows) return;
    const float* rp = x + (size_t)row * d;
    float s = 0.0f;
    for (int i = lane; i < d; i += 32) {
        float v = rp[i];
        s += v * v;
    }
    #pragma unroll
    for (int off = 16; off > 0; off >>= 1)
        s += __shfl_xor(s, off, 32);
    if (lane == 0) out[row] = sqrtf(s);
}

// Build a 16-element bf16 fragment from two 8-element LDS groups (2x ds_load_b128).
static __device__ inline v16bf make_frag(const __bf16* p0, const __bf16* p1) {
    v8bf a = *reinterpret_cast<const v8bf*>(p0);
    v8bf b = *reinterpret_cast<const v8bf*>(p1);
    return __builtin_shufflevector(a, b, 0,1,2,3,4,5,6,7,8,9,10,11,12,13,14,15);
}

// Split 16 f32 into bf16 hi/lo, store as two b128 groups each into LDS.
static __device__ inline void split_store(const f32x4 g[4], __bf16* hiDst, __bf16* loDst) {
    float v[16];
    #pragma unroll
    for (int q = 0; q < 4; ++q) {
        v[4*q+0] = g[q].x; v[4*q+1] = g[q].y; v[4*q+2] = g[q].z; v[4*q+3] = g[q].w;
    }
    v8bf h0, h1, l0, l1;
    #pragma unroll
    for (int e = 0; e < 8; ++e) {
        __bf16 hi = (__bf16)v[e];
        h0[e] = hi;
        l0[e] = (__bf16)(v[e] - (float)hi);
    }
    #pragma unroll
    for (int e = 0; e < 8; ++e) {
        __bf16 hi = (__bf16)v[8 + e];
        h1[e] = hi;
        l1[e] = (__bf16)(v[8 + e] - (float)hi);
    }
    *reinterpret_cast<v8bf*>(hiDst)     = h0;
    *reinterpret_cast<v8bf*>(hiDst + 8) = h1;
    *reinterpret_cast<v8bf*>(loDst)     = l0;
    *reinterpret_cast<v8bf*>(loDst + 8) = l1;
}

// ---------------------------------------------------------------------------
// Cosine-similarity GEMM: out[m,n] = dot(A[m,:],B[n,:]) / max(n1[m]*n2[n],eps)
// Split-bf16 (hi/lo) WMMA, f32 accumulation. Conversion done ONCE per element
// at staging time; LDS holds bf16 hi/lo tiles. Software-pipelined global loads.
// Block tile 128x128, 8 waves (2Mx4N), wave tile 64x32 = 4x2 of 16x16.
// ---------------------------------------------------------------------------
__launch_bounds__(256, 1)
__global__ void cossim_gemm_kernel(const float* __restrict__ A,   // [Mrows, D]
                                   const float* __restrict__ B,   // [Nrows, D]
                                   const float* __restrict__ n1,  // [Mrows]
                                   const float* __restrict__ n2,  // [Nrows]
                                   float* __restrict__ out,       // [Mrows, Nrows]
                                   int D, int Ncols) {
    __shared__ __align__(16) __bf16 sAhi[BM * LDSW];
    __shared__ __align__(16) __bf16 sAlo[BM * LDSW];
    __shared__ __align__(16) __bf16 sBhi[BN * LDSW];
    __shared__ __align__(16) __bf16 sBlo[BN * LDSW];

    const int tid  = threadIdx.x;
    const int wave = tid >> 5;
    const int lane = tid & 31;
    const int h    = lane >> 4;    // lane half
    const int ln   = lane & 15;

    const int waveM  = (wave & 1) * 64;
    const int waveN  = (wave >> 1) * 32;
    const int blockM = blockIdx.y * BM;
    const int blockN = blockIdx.x * BN;

    // staging ownership: 2 threads per row, 16 contiguous floats each
    const int srow  = tid >> 1;          // 0..127
    const int shalf = (tid & 1) * 16;    // col base 0 or 16

    v8f acc[4][2];
    const v8f vzero = {0.f,0.f,0.f,0.f,0.f,0.f,0.f,0.f};
    #pragma unroll
    for (int mt = 0; mt < 4; ++mt)
        #pragma unroll
        for (int nt = 0; nt < 2; ++nt)
            acc[mt][nt] = vzero;

    // A-frag K-group column bases (contiguous K order, matches B):
    // h==0: elems 0-7 = K 0..7,   elems 8-15 = K 8..15
    // h==1: elems 0-7 = K 16..23, elems 8-15 = K 24..31
    const int c0 = h ? 16 : 0;
    const int c1 = h ? 24 : 8;

    // ---- prologue: global loads for k0 = 0 ----
    f32x4 ga[4], gb[4];
    {
        const float* ap = A + (size_t)(blockM + srow) * D + shalf;
        const float* bp = B + (size_t)(blockN + srow) * D + shalf;
        #pragma unroll
        for (int q = 0; q < 4; ++q) {
            ga[q] = *reinterpret_cast<const f32x4*>(ap + 4 * q);
            gb[q] = *reinterpret_cast<const f32x4*>(bp + 4 * q);
        }
    }

    for (int k0 = 0; k0 < D; k0 += TILE_K) {
        __syncthreads();   // previous compute finished reading LDS

        // ---- convert current tile regs -> bf16 hi/lo, store to LDS ----
        split_store(ga, &sAhi[srow * LDSW + shalf], &sAlo[srow * LDSW + shalf]);
        split_store(gb, &sBhi[srow * LDSW + shalf], &sBlo[srow * LDSW + shalf]);

        // ---- issue next tile's global loads (overlap with compute below) ----
        if (k0 + TILE_K < D) {
            const float* ap = A + (size_t)(blockM + srow) * D + k0 + TILE_K + shalf;
            const float* bp = B + (size_t)(blockN + srow) * D + k0 + TILE_K + shalf;
            #pragma unroll
            for (int q = 0; q < 4; ++q) {
                ga[q] = *reinterpret_cast<const f32x4*>(ap + 4 * q);
                gb[q] = *reinterpret_cast<const f32x4*>(bp + 4 * q);
            }
            // cache-hierarchy hint for the tile after next (global_prefetch_b8)
            if (k0 + 2 * TILE_K < D) {
                __builtin_prefetch(A + (size_t)(blockM + srow) * D + k0 + 2 * TILE_K + shalf, 0, 0);
                __builtin_prefetch(B + (size_t)(blockN + srow) * D + k0 + 2 * TILE_K + shalf, 0, 0);
            }
        }

        __syncthreads();

        // ---- B fragments (32x16 KxN): element e = K (h*16 + e), N = ln ----
        v16bf bhi[2], blo[2];
        #pragma unroll
        for (int nt = 0; nt < 2; ++nt) {
            const int rbase = (waveN + nt * 16 + ln) * LDSW + h * 16;
            bhi[nt] = make_frag(&sBhi[rbase], &sBhi[rbase + 8]);
            blo[nt] = make_frag(&sBlo[rbase], &sBlo[rbase + 8]);
        }

        // ---- per M-tile: A fragments + 3 WMMAs per (mt,nt) ----
        #pragma unroll
        for (int mt = 0; mt < 4; ++mt) {
            const int rbase = (waveM + mt * 16 + ln) * LDSW;
            v16bf ahi = make_frag(&sAhi[rbase + c0], &sAhi[rbase + c1]);
            v16bf alo = make_frag(&sAlo[rbase + c0], &sAlo[rbase + c1]);
            #pragma unroll
            for (int nt = 0; nt < 2; ++nt) {
                acc[mt][nt] = __builtin_amdgcn_wmma_f32_16x16x32_bf16(
                    false, ahi, false, bhi[nt], (short)0, acc[mt][nt], false, false);
                acc[mt][nt] = __builtin_amdgcn_wmma_f32_16x16x32_bf16(
                    false, ahi, false, blo[nt], (short)0, acc[mt][nt], false, false);
                acc[mt][nt] = __builtin_amdgcn_wmma_f32_16x16x32_bf16(
                    false, alo, false, bhi[nt], (short)0, acc[mt][nt], false, false);
            }
        }
    }

    // ---- epilogue: divide by max(n1[m]*n2[n], eps), write out ----
    float n2v[2];
    #pragma unroll
    for (int nt = 0; nt < 2; ++nt)
        n2v[nt] = n2[blockN + waveN + nt * 16 + ln];

    #pragma unroll
    for (int mt = 0; mt < 4; ++mt) {
        float n1v[8];
        #pragma unroll
        for (int r = 0; r < 8; ++r)
            n1v[r] = n1[blockM + waveM + mt * 16 + r + 8 * h];
        #pragma unroll
        for (int nt = 0; nt < 2; ++nt) {
            const int col = blockN + waveN + nt * 16 + ln;
            #pragma unroll
            for (int r = 0; r < 8; ++r) {
                const int row = blockM + waveM + mt * 16 + r + 8 * h;
                float denom = fmaxf(n1v[r] * n2v[nt], 1e-8f);
                out[(size_t)row * Ncols + col] = acc[mt][nt][r] / denom;
            }
        }
    }
}

// ---------------------------------------------------------------------------
extern "C" void kernel_launch(void* const* d_in, const int* in_sizes, int n_in,
                              void* d_out, int out_size, void* d_ws, size_t ws_size,
                              hipStream_t stream) {
    const float* in1 = (const float*)d_in[0];
    const float* in2 = (const float*)d_in[1];
    float* out = (float*)d_out;

    const int D     = 512;
    const int Mrows = in_sizes[0] / D;   // 8192
    const int Nrows = in_sizes[1] / D;   // 8192

    float* n1 = (float*)d_ws;            // [Mrows]
    float* n2 = n1 + Mrows;              // [Nrows]

    {
        int threads = 256;
        int blocks1 = (Mrows * 32 + threads - 1) / threads;
        int blocks2 = (Nrows * 32 + threads - 1) / threads;
        rownorm_kernel<<<blocks1, threads, 0, stream>>>(in1, n1, Mrows, D);
        rownorm_kernel<<<blocks2, threads, 0, stream>>>(in2, n2, Nrows, D);
    }

    dim3 grid(Nrows / BN, Mrows / BM);
    cossim_gemm_kernel<<<grid, 256, 0, stream>>>(in1, in2, n1, n2, out, D, Nrows);
}
